// AttentionMessagePassing_1872605741888
// MI455X (gfx1250) — matrix operand order repaired and checked by reading, not compiled, not run on hardware
//
#include <hip/hip_runtime.h>
#include <math.h>

typedef __attribute__((ext_vector_type(16))) _Float16 v16h;
typedef __attribute__((ext_vector_type(8)))  float    v8f;

#define B_     8
#define NN     128
#define FND    64
#define FED    64
#define JT     16            // j tile per wave (WMMA M)
#define IC     8             // i chunk per wave
#define ISPLIT (NN / IC)     // 16
#define NJT    (NN / JT)     // 8

__device__ __forceinline__ v8f wmma_f16(v16h a, v16h b, v8f c) {
  return __builtin_amdgcn_wmma_f32_16x16x32_f16(false, a, false, b, (short)0, c, false, false);
}

// Pack 16 f32 (two runs of 8 consecutive K values) into one f16 operand register set.
__device__ __forceinline__ v16h pack16(float4 t0, float4 t1, float4 t2, float4 t3) {
  v16h a;
  a[0]  = (_Float16)t0.x; a[1]  = (_Float16)t0.y; a[2]  = (_Float16)t0.z; a[3]  = (_Float16)t0.w;
  a[4]  = (_Float16)t1.x; a[5]  = (_Float16)t1.y; a[6]  = (_Float16)t1.z; a[7]  = (_Float16)t1.w;
  a[8]  = (_Float16)t2.x; a[9]  = (_Float16)t2.y; a[10] = (_Float16)t2.z; a[11] = (_Float16)t2.w;
  a[12] = (_Float16)t3.x; a[13] = (_Float16)t3.y; a[14] = (_Float16)t3.z; a[15] = (_Float16)t3.w;
  return a;
}

// A operand pair (K=0..31 and K=32..63) for a 16x64 row-major f32 tile with ld=64.
// ISA A layout: lane holds row M=lane%16; per k-step the lane's 16 values are
// K = half*8 + [0..7] and K = 16 + half*8 + [0..7]  ->  4 x b128 loads per operand.
__device__ __forceinline__ void load_a_pair(const float* __restrict__ tile, int lane,
                                            v16h& a0, v16h& a1) {
  const int row  = lane & 15;
  const int half = lane >> 4;
  const float4* q = (const float4*)(tile + row * 64 + half * 8);
  float4 t0 = q[0],  t1 = q[1];        // K = half*8 + 0..7
  float4 t2 = q[4],  t3 = q[5];        // K = 16 + half*8 + 0..7
  float4 u0 = q[8],  u1 = q[9];        // K = 32 + half*8 + 0..7
  float4 u2 = q[12], u3 = q[13];       // K = 48 + half*8 + 0..7
  a0 = pack16(t0, t1, t2, t3);
  a1 = pack16(u0, u1, u2, u3);
}

// Coalesced stage of a 64x64 f32 W block into LDS (32 x b128 per lane).
__device__ __forceinline__ void stage_w(float* __restrict__ wst,
                                        const float* __restrict__ Wblk, int lane) {
  const float4* s = (const float4*)Wblk;
  float4* d = (float4*)wst;
#pragma unroll
  for (int t = 0; t < 32; ++t) d[t * 32 + lane] = s[t * 32 + lane];
}

// B operand: 32x16 f16 tile at (kb, n0) built from the LDS-staged 64x64 W block.
// ISA B layout: lane holds column N = n0 + lane%16; elements 0-7 K=kb+half*8+0..7, elements 8-15 K=kb+16+half*8+0..7.
__device__ __forceinline__ v16h build_b(const float* __restrict__ wst, int kb, int n0, int lane) {
  const int col = n0 + (lane & 15);
  const int kh  = (lane >> 4) * 8;
  const float* p = wst + (kb + kh) * 64 + col;
  v16h b;
#pragma unroll
  for (int v = 0; v < 8; ++v) {
    b[2 * v]     = (_Float16)p[(2 * v + (v >= 4 ? 8 : 0)) * 64];
    b[2 * v + 1] = (_Float16)p[(2 * v + 1 + (v >= 4 ? 8 : 0)) * 64];
  }
  return b;
}

__global__ void zero_out_kernel(float* __restrict__ out, int n) {
  int t = blockIdx.x * blockDim.x + threadIdx.x;
  if (t < n) out[t] = 0.0f;
}

__global__ __launch_bounds__(32)
void gated_mp_kernel(const float* __restrict__ H,     // (B, N, FN)
                     const float* __restrict__ Aadj,  // (B, N, N)
                     const float* __restrict__ E,     // (B, N, N, FE)
                     const float* __restrict__ Watt,  // (2FN+FE, FN)
                     const float* __restrict__ Wnei,  // (2FN+FE, FN)
                     const float* __restrict__ batt,  // (FN)
                     const float* __restrict__ bnei,  // (FN)
                     float* __restrict__ out) {       // (B, N, FN)
  __shared__ float wst[64 * 64];            // 16 KB W staging (reused 6x)
  __shared__ float Patt[16][FND];           // 4 KB (16-row aligned i-chunk)
  __shared__ float Pnei[16][FND];           // 4 KB

  const int lane = threadIdx.x;             // 0..31
  const int col  = lane & 15;
  const int half = lane >> 4;

  const int bid = blockIdx.x;
  const int b   = bid / (NJT * ISPLIT);
  const int rem = bid % (NJT * ISPLIT);
  const int j0  = (rem / ISPLIT) * JT;
  const int i0  = (rem % ISPLIT) * IC;
  const int i0p = i0 & ~15;                 // 16-aligned anchor for the P tile
  const int ioff = i0 - i0p;                // 0 or 8

  const float* Hb = H + (size_t)b * NN * FND;

  // ---------- setup: stage W blocks through LDS, build operands, run small WMMAs ----------

  // E-branch B operands (persistent across the i-loop): [kstep][ntile]
  v16h BeA[2][4], BeN[2][4];
  stage_w(wst, Watt + 2 * FND * FND, lane);  // We_att
  __syncthreads();
#pragma unroll
  for (int ks = 0; ks < 2; ++ks)
#pragma unroll
    for (int nt = 0; nt < 4; ++nt) BeA[ks][nt] = build_b(wst, ks * 32, nt * 16, lane);
  __syncthreads();
  stage_w(wst, Wnei + 2 * FND * FND, lane);  // We_nei
  __syncthreads();
#pragma unroll
  for (int ks = 0; ks < 2; ++ks)
#pragma unroll
    for (int nt = 0; nt < 4; ++nt) BeN[ks][nt] = build_b(wst, ks * 32, nt * 16, lane);
  __syncthreads();

  // Q tiles (column term, rows j0..j0+15), kept in registers in C/D layout.
  v8f Qa[4], Qn[4];
  v16h hj0, hj1, hi0, hi1;
  load_a_pair(Hb + (size_t)j0 * FND, lane, hj0, hj1);
  load_a_pair(Hb + (size_t)i0p * FND, lane, hi0, hi1);

  stage_w(wst, Watt + FND * FND, lane);      // Wj_att
  __syncthreads();
#pragma unroll
  for (int nt = 0; nt < 4; ++nt) {
    v8f c = {};
    c = wmma_f16(hj0, build_b(wst, 0,  nt * 16, lane), c);
    c = wmma_f16(hj1, build_b(wst, 32, nt * 16, lane), c);
    Qa[nt] = c;
  }
  __syncthreads();
  stage_w(wst, Wnei + FND * FND, lane);      // Wj_nei
  __syncthreads();
#pragma unroll
  for (int nt = 0; nt < 4; ++nt) {
    v8f c = {};
    c = wmma_f16(hj0, build_b(wst, 0,  nt * 16, lane), c);
    c = wmma_f16(hj1, build_b(wst, 32, nt * 16, lane), c);
    Qn[nt] = c;
  }
  __syncthreads();

  // P tiles (row term, rows i0p..i0p+15) -> LDS for broadcast reads in the i-loop.
  stage_w(wst, Watt, lane);                  // Wi_att
  __syncthreads();
#pragma unroll
  for (int nt = 0; nt < 4; ++nt) {
    v8f c = {};
    c = wmma_f16(hi0, build_b(wst, 0,  nt * 16, lane), c);
    c = wmma_f16(hi1, build_b(wst, 32, nt * 16, lane), c);
#pragma unroll
    for (int r = 0; r < 8; ++r) Patt[r + half * 8][nt * 16 + col] = c[r];
  }
  __syncthreads();
  stage_w(wst, Wnei, lane);                  // Wi_nei
  __syncthreads();
#pragma unroll
  for (int nt = 0; nt < 4; ++nt) {
    v8f c = {};
    c = wmma_f16(hi0, build_b(wst, 0,  nt * 16, lane), c);
    c = wmma_f16(hi1, build_b(wst, 32, nt * 16, lane), c);
#pragma unroll
    for (int r = 0; r < 8; ++r) Pnei[r + half * 8][nt * 16 + col] = c[r];
  }
  __syncthreads();

  float ba[4], bn[4];
#pragma unroll
  for (int nt = 0; nt < 4; ++nt) {
    ba[nt] = batt[nt * 16 + col];
    bn[nt] = bnei[nt * 16 + col];
  }

  // ---------- main loop over the i-chunk: software-pipelined E tiles ----------

  const float* E0 = E + (((size_t)b * NN + i0) * NN + j0) * FED;   // tile il at E0 + il*NN*FED
  const float* Ab = Aadj + ((size_t)b * NN + i0) * NN + j0 + half * 8;

  v8f oacc[4] = {v8f{}, v8f{}, v8f{}, v8f{}};

  v16h ea0, ea1;
  load_a_pair(E0, lane, ea0, ea1);

  for (int il = 0; il < IC; ++il) {
    // prefetch il+2 into L2 (global_prefetch_b8; each lane covers a slice of the 4KB tile)
    if (il + 2 < IC)
      __builtin_prefetch(E0 + (size_t)(il + 2) * NN * FED + lane * 32, 0, 3);

    // pipeline: issue next tile's loads before consuming the current tile
    const int nx = (il + 1 < IC) ? il + 1 : il;
    v16h na0, na1;
    load_a_pair(E0 + (size_t)nx * NN * FED, lane, na0, na1);

    // A[b, i, j0+half*8 .. +7]: 8 contiguous floats per lane -> 2 x b128 loads
    const float4* ap = (const float4*)(Ab + (size_t)il * NN);
    const float4 av0 = ap[0], av1 = ap[1];
    const float av[8] = {av0.x, av0.y, av0.z, av0.w, av1.x, av1.y, av1.z, av1.w};

#pragma unroll
    for (int nt = 0; nt < 4; ++nt) {
      v8f ca = {};
      ca = wmma_f16(ea0, BeA[0][nt], ca);
      ca = wmma_f16(ea1, BeA[1][nt], ca);
      v8f cn = {};
      cn = wmma_f16(ea0, BeN[0][nt], cn);
      cn = wmma_f16(ea1, BeN[1][nt], cn);

      const float pa = Patt[il + ioff][nt * 16 + col];
      const float pn = Pnei[il + ioff][nt * 16 + col];
#pragma unroll
      for (int r = 0; r < 8; ++r) {
        const float aij = av[r];
        const float att = aij * (ca[r] + pa + Qa[nt][r]) + ba[nt];
        const float cnv = aij * (cn[r] + pn + Qn[nt][r]) + bn[nt];
        const float sig = 1.0f / (1.0f + __expf(-att));
        const float rel = cnv > 0.0f ? cnv : 0.0f;
        oacc[nt][r] += sig * rel;
      }
    }
    ea0 = na0;
    ea1 = na1;
  }

  // ---------- accumulate partials (over i-chunks) into out with f32 global atomics ----------
#pragma unroll
  for (int nt = 0; nt < 4; ++nt)
#pragma unroll
    for (int r = 0; r < 8; ++r) {
      const int j = j0 + r + half * 8;
      atomicAdd(&out[((size_t)b * NN + j) * FND + nt * 16 + col], oacc[nt][r]);
    }
}

extern "C" void kernel_launch(void* const* d_in, const int* in_sizes, int n_in,
                              void* d_out, int out_size, void* d_ws, size_t ws_size,
                              hipStream_t stream) {
  const float* H    = (const float*)d_in[0];
  const float* Aadj = (const float*)d_in[1];
  const float* E    = (const float*)d_in[2];
  const float* Watt = (const float*)d_in[3];
  const float* Wnei = (const float*)d_in[4];
  const float* batt = (const float*)d_in[5];
  const float* bnei = (const float*)d_in[6];
  float* out = (float*)d_out;

  // out must start at zero (atomics accumulate); harness poisons d_out.
  zero_out_kernel<<<(out_size + 255) / 256, 256, 0, stream>>>(out, out_size);

  const int nblocks = B_ * NJT * ISPLIT;   // 8 * 8 * 16 = 1024 waves
  gated_mp_kernel<<<nblocks, 32, 0, stream>>>(H, Aadj, E, Watt, Wnei, batt, bnei, out);
}
